// MultiHeadAttention_38371237822938
// MI455X (gfx1250) — compile-verified, matrix operand order checked by reading, not run on hardware
//
#include <hip/hip_runtime.h>
#include <hip/hip_bf16.h>

// Problem sizes (fixed by reference)
static constexpr int CB  = 2;     // batch
static constexpr int CS  = 2048;  // seq
static constexpr int CD  = 1024;  // model dim
static constexpr int CH  = 16;    // heads
static constexpr int CDH = 64;    // head dim

typedef __attribute__((ext_vector_type(16))) __bf16 v16bf;
typedef __attribute__((ext_vector_type(8)))  __bf16 v8bf;
typedef __attribute__((ext_vector_type(8)))  float  v8f;

// ---------- bf16 helpers (round-to-nearest-even via integer path) ----------
__device__ __forceinline__ unsigned short f2bfu(float f) {
    union { float f; unsigned u; } v; v.f = f;
    unsigned r = v.u + 0x7FFFu + ((v.u >> 16) & 1u);
    return (unsigned short)(r >> 16);
}
__device__ __forceinline__ __bf16 us2bf(unsigned short h) {
    __bf16 b; __builtin_memcpy(&b, &h, 2); return b;
}
__device__ __forceinline__ __bf16 f2bf(float f) { return us2bf(f2bfu(f)); }

// ---------- WMMA fragment loaders ----------
// 16-bit A fragment (16x32, M x K): lane 0-15 -> row=lane, koff=0 ; lane 16-31 -> row=lane-16, koff=8.
// VGPRs 0-3 hold K = koff+0..7 (contiguous 16B), VGPRs 4-7 hold K = 16+koff+0..7 (contiguous 16B).
// B fragment (32x16, K x N) mirrors this with N in place of M, so loading from a
// row-major [N,K] (i.e. transposed-B) buffer uses the identical pattern.
__device__ __forceinline__ const __bf16* frag_ptr(const __bf16* base, int idx_base,
                                                  long ld, int kbase, int lane) {
    int idx  = lane & 15;
    int koff = (lane >> 4) << 3;
    return base + (long)(idx_base + idx) * ld + kbase + koff;
}
__device__ __forceinline__ v16bf load_frag(const __bf16* p) {
    v8bf lo = *(const v8bf*)(p);
    v8bf hi = *(const v8bf*)(p + 16);
    return __builtin_shufflevector(lo, hi, 0,1,2,3,4,5,6,7,8,9,10,11,12,13,14,15);
}
__device__ __forceinline__ v16bf load_frag_lds(const unsigned short* base, int ld,
                                               int kbase, int lane) {
    int idx  = lane & 15;
    int koff = (lane >> 4) << 3;
    const unsigned short* p = base + idx * ld + kbase + koff;
    v8bf lo = *(const v8bf*)(p);
    v8bf hi = *(const v8bf*)(p + 16);
    return __builtin_shufflevector(lo, hi, 0,1,2,3,4,5,6,7,8,9,10,11,12,13,14,15);
}
__device__ __forceinline__ v8f wmma_bf16(v16bf a, v16bf b, v8f c) {
    return __builtin_amdgcn_wmma_f32_16x16x32_bf16(false, a, false, b, (short)0, c, false, false);
}

// ---------- CDNA5 async copy to LDS (ASYNCcnt-tracked, no VGPR round trip) ----------
// lds_off is a byte offset into the workgroup's LDS allocation (our single
// __shared__ block starts at offset 0).
__device__ __forceinline__ void async_load_b128(unsigned lds_off, const void* gaddr) {
    asm volatile("global_load_async_to_lds_b128 %0, %1, off"
                 :: "v"(lds_off), "v"(gaddr) : "memory");
}
__device__ __forceinline__ void wait_async0() {
    asm volatile("s_wait_asynccnt 0x0" ::: "memory");
}

// ---------- staging kernels ----------
__global__ __launch_bounds__(256) void k_cvt(const float* __restrict__ in,
                                             __bf16* __restrict__ out, int n) {
    int i = blockIdx.x * 256 + threadIdx.x;
    if (i < n) out[i] = f2bf(in[i]);
}

// W [H, D, DH] fp32 -> Wt [H, DH, D] bf16
__global__ __launch_bounds__(256) void k_tr_hde(const float* __restrict__ w,
                                                __bf16* __restrict__ wt) {
    long i = (long)blockIdx.x * 256 + threadIdx.x;
    long total = (long)CH * CD * CDH;
    if (i >= total) return;
    int e = (int)(i % CDH);
    long t = i / CDH;
    int d = (int)(t % CD);
    int h = (int)(t / CD);
    wt[((long)h * CDH + e) * CD + d] = f2bf(w[i]);
}

// Wo [D, D] fp32 (K x N) -> Wot [N, K] bf16
__global__ __launch_bounds__(256) void k_tr_dd(const float* __restrict__ w,
                                               __bf16* __restrict__ wt) {
    long i = (long)blockIdx.x * 256 + threadIdx.x;
    long total = (long)CD * CD;
    if (i >= total) return;
    int n = (int)(i % CD);
    int k = (int)(i / CD);
    wt[(long)n * CD + k] = f2bf(w[i]);
}

// ---------- QKV projection: per wave, 16 rows x DH=64 cols of Q,K,V ----------
// Q,K stored [BH, S, DH] row-major bf16; V stored transposed [BH, DH, S] bf16.
__global__ __launch_bounds__(256) void k_qkv(const __bf16* __restrict__ Xb,
                                             const __bf16* __restrict__ Wqt,
                                             const __bf16* __restrict__ Wkt,
                                             const __bf16* __restrict__ Wvt,
                                             __bf16* __restrict__ Q,
                                             __bf16* __restrict__ K,
                                             __bf16* __restrict__ Vt) {
    const int lane   = threadIdx.x & 31;
    const int wv     = threadIdx.x >> 5;
    const int linear = blockIdx.x * 8 + wv;      // 4096 waves total
    const int strip  = linear & 127;             // S/16 = 128 strips
    const int bh     = linear >> 7;              // 0..31
    const int b      = bh >> 4;
    const int h      = bh & 15;

    const __bf16* X   = Xb + ((long)b * CS + strip * 16) * CD;
    const __bf16* wq  = Wqt + (long)h * CDH * CD;
    const __bf16* wk  = Wkt + (long)h * CDH * CD;
    const __bf16* wvp = Wvt + (long)h * CDH * CD;

    v8f accq[4] = {}; v8f acck[4] = {}; v8f accv[4] = {};

    for (int d0 = 0; d0 < CD; d0 += 32) {
        v16bf af = load_frag(frag_ptr(X, 0, CD, d0, lane));
#pragma unroll
        for (int nt = 0; nt < 4; ++nt) {
            v16bf bq = load_frag(frag_ptr(wq,  nt * 16, CD, d0, lane));
            accq[nt] = wmma_bf16(af, bq, accq[nt]);
            v16bf bk = load_frag(frag_ptr(wk,  nt * 16, CD, d0, lane));
            acck[nt] = wmma_bf16(af, bk, acck[nt]);
            v16bf bv = load_frag(frag_ptr(wvp, nt * 16, CD, d0, lane));
            accv[nt] = wmma_bf16(af, bv, accv[nt]);
        }
    }

    // C/D layout: lane 0-15 -> col=lane, rows 0..7 ; lane 16-31 -> col=lane-16, rows 8..15
    const int col   = lane & 15;
    const int rbase = (lane >> 4) << 3;
    const long qk_base = ((long)bh * CS + strip * 16) * CDH;
    const long vt_base = (long)bh * CDH * CS;
#pragma unroll
    for (int nt = 0; nt < 4; ++nt) {
        const int c = nt * 16 + col;
#pragma unroll
        for (int i = 0; i < 8; ++i) {
            const int r = rbase + i;
            Q[qk_base + (long)r * CDH + c] = f2bf(accq[nt][i]);
            K[qk_base + (long)r * CDH + c] = f2bf(acck[nt][i]);
            Vt[vt_base + (long)c * CS + strip * 16 + r] = f2bf(accv[nt][i]);
        }
    }
}

// ---------- fused flash attention ----------
// Per wave: 16 queries x all keys. Per workgroup (8 waves, same (b,h)): K/V
// tiles staged once into LDS with async b128 copies, double buffered so the
// next tile's copy overlaps the current tile's WMMAs.
static constexpr int KVLD       = 72;                       // ushorts per LDS row (pad: 144B)
static constexpr int TILE_BYTES = 64 * KVLD * 2;            // 9216 per K or V tile
static constexpr int BUF_BYTES  = 2 * TILE_BYTES;           // K+V per stage
static constexpr int PLDS_OFF   = 2 * BUF_BYTES;            // double buffer end
static constexpr int SMEM_TOTAL = PLDS_OFF + 8 * 16 * KVLD * 2;

__global__ __launch_bounds__(256) void k_attn(const __bf16* __restrict__ Q,
                                              const __bf16* __restrict__ Km,
                                              const __bf16* __restrict__ Vt,
                                              __bf16* __restrict__ Cc) {
    __shared__ __align__(16) unsigned char smem[SMEM_TOTAL];

    const int tid    = threadIdx.x;
    const int lane   = tid & 31;
    const int wv     = tid >> 5;
    const int linear = blockIdx.x * 8 + wv;
    const int strip  = linear & 127;
    const int bh     = linear >> 7;   // all 8 waves of a block share bh (128 strips/bh)
    const int b      = bh >> 4;
    const int h      = bh & 15;

    const __bf16* q    = Q  + ((long)bh * CS + strip * 16) * CDH;
    const __bf16* kmat = Km + (long)bh * CS * CDH;
    const __bf16* vt   = Vt + (long)bh * CDH * CS;

    // async-stage K tile (contiguous 8KB slab of Km) and V tile (64 rows of
    // 128B from Vt[DH,S]) into padded LDS rows; 4 b128 copies per thread.
    auto stage = [&](int bufsel, int kt) {
        const unsigned base = (unsigned)(bufsel * BUF_BYTES);
        const char* kg = (const char*)kmat + (size_t)kt * CDH * 2;
#pragma unroll
        for (int j = 0; j < 2; ++j) {
            int c = tid + j * 256;            // 512 chunks of 16B
            int row = c >> 3, bo = (c & 7) * 16;
            async_load_b128(base + row * (KVLD * 2) + bo,
                            kg + (size_t)row * (CDH * 2) + bo);
        }
#pragma unroll
        for (int j = 0; j < 2; ++j) {
            int c = tid + j * 256;
            int row = c >> 3, bo = (c & 7) * 16;
            async_load_b128(base + TILE_BYTES + row * (KVLD * 2) + bo,
                            (const char*)vt + ((size_t)row * CS + kt) * 2 + bo);
        }
    };

    // Q fragments for K-dim = DH = 64 -> two 16x32 A fragments, loaded once
    v16bf qf0 = load_frag(frag_ptr(q, 0, CDH, 0,  lane));
    v16bf qf1 = load_frag(frag_ptr(q, 0, CDH, 32, lane));

    v8f acc[4] = {};
    float m_s[8], l_s[8];
#pragma unroll
    for (int i = 0; i < 8; ++i) { m_s[i] = -3.0e38f; l_s[i] = 0.0f; }

    const int col   = lane & 15;
    const int rbase = (lane >> 4) << 3;
    unsigned short* plw = (unsigned short*)(smem + PLDS_OFF) + wv * 16 * KVLD;

    stage(0, 0);

    for (int kt = 0; kt < CS; kt += 64) {
        const int cur = (kt >> 6) & 1;
        wait_async0();        // this wave's staged copies have landed
        __syncthreads();      // all waves' copies have landed; prior reads done
        if (kt + 64 < CS) stage(cur ^ 1, kt + 64);

        const unsigned short* kbuf = (const unsigned short*)(smem + cur * BUF_BYTES);
        const unsigned short* vbuf = kbuf + TILE_BYTES / 2;

        // scores tile: 16 queries x 64 keys (K fragments from LDS)
        v8f s[4];
#pragma unroll
        for (int nt = 0; nt < 4; ++nt) {
            v16bf kb0 = load_frag_lds(kbuf + nt * 16 * KVLD, KVLD, 0,  lane);
            v16bf kb1 = load_frag_lds(kbuf + nt * 16 * KVLD, KVLD, 32, lane);
            v8f z = {};
            z = wmma_bf16(qf0, kb0, z);
            z = wmma_bf16(qf1, kb1, z);
            s[nt] = z;
        }

        // online softmax: a row's 16 cols of a tile live in one 16-lane half,
        // so shfl_xor within width-16 reduces a row.
        const float sc = 0.125f;  // 1/sqrt(64)
#pragma unroll
        for (int i = 0; i < 8; ++i) {
            float a0 = s[0][i] * sc, a1 = s[1][i] * sc,
                  a2 = s[2][i] * sc, a3 = s[3][i] * sc;
            float mx = fmaxf(fmaxf(a0, a1), fmaxf(a2, a3));
            mx = fmaxf(mx, __shfl_xor(mx, 1, 16));
            mx = fmaxf(mx, __shfl_xor(mx, 2, 16));
            mx = fmaxf(mx, __shfl_xor(mx, 4, 16));
            mx = fmaxf(mx, __shfl_xor(mx, 8, 16));
            float mnew = fmaxf(m_s[i], mx);
            float f    = __expf(m_s[i] - mnew);
            m_s[i] = mnew;
            float e0 = __expf(a0 - mnew), e1 = __expf(a1 - mnew),
                  e2 = __expf(a2 - mnew), e3 = __expf(a3 - mnew);
            s[0][i] = e0; s[1][i] = e1; s[2][i] = e2; s[3][i] = e3;
            float rs = e0 + e1 + e2 + e3;
            rs += __shfl_xor(rs, 1, 16);
            rs += __shfl_xor(rs, 2, 16);
            rs += __shfl_xor(rs, 4, 16);
            rs += __shfl_xor(rs, 8, 16);
            l_s[i] = l_s[i] * f + rs;
            acc[0][i] *= f; acc[1][i] *= f; acc[2][i] *= f; acc[3][i] *= f;
        }

        // repack P (C layout) -> LDS -> A fragments (same-wave DS ops in order)
#pragma unroll
        for (int nt = 0; nt < 4; ++nt)
#pragma unroll
            for (int i = 0; i < 8; ++i)
                plw[(rbase + i) * KVLD + nt * 16 + col] = f2bfu(s[nt][i]);

        v16bf pf0 = load_frag_lds(plw, KVLD, 0,  lane);
        v16bf pf1 = load_frag_lds(plw, KVLD, 32, lane);

        // acc += P(16x64) x V(64x64 slice); V fragments from LDS
#pragma unroll
        for (int nt = 0; nt < 4; ++nt) {
            v16bf vb0 = load_frag_lds(vbuf + nt * 16 * KVLD, KVLD, 0,  lane);
            v16bf vb1 = load_frag_lds(vbuf + nt * 16 * KVLD, KVLD, 32, lane);
            acc[nt] = wmma_bf16(pf0, vb0, acc[nt]);
            acc[nt] = wmma_bf16(pf1, vb1, acc[nt]);
        }
    }

    // epilogue: normalize and store into concat layout [B,S,H*DH] bf16
    float inv[8];
#pragma unroll
    for (int i = 0; i < 8; ++i) inv[i] = 1.0f / l_s[i];
#pragma unroll
    for (int nt = 0; nt < 4; ++nt) {
        const int c = h * CDH + nt * 16 + col;
#pragma unroll
        for (int i = 0; i < 8; ++i) {
            const int r = strip * 16 + rbase + i;
            Cc[((long)b * CS + r) * CD + c] = f2bf(acc[nt][i] * inv[i]);
        }
    }
}

// ---------- output projection: out[4096,1024] = Cc x Wo (fp32 out) ----------
__global__ __launch_bounds__(256) void k_out(const __bf16* __restrict__ Cc,
                                             const __bf16* __restrict__ Wot,
                                             float* __restrict__ out) {
    const int lane   = threadIdx.x & 31;
    const int wv     = threadIdx.x >> 5;
    const int linear = blockIdx.x * 8 + wv;      // 4096 waves = (4096/16) * (1024/64)
    const int ntile  = linear & 15;              // 16 n-tiles of 64
    const int mstrip = linear >> 4;              // 256 m-strips of 16

    const __bf16* a  = Cc  + (long)mstrip * 16 * CD;
    const __bf16* bw = Wot + (long)ntile * 64 * CD;

    v8f acc[4] = {};
    for (int k0 = 0; k0 < CD; k0 += 32) {
        v16bf af = load_frag(frag_ptr(a, 0, CD, k0, lane));
#pragma unroll
        for (int nt = 0; nt < 4; ++nt) {
            v16bf bf_ = load_frag(frag_ptr(bw, nt * 16, CD, k0, lane));
            acc[nt] = wmma_bf16(af, bf_, acc[nt]);
        }
    }

    const int col   = lane & 15;
    const int rbase = (lane >> 4) << 3;
#pragma unroll
    for (int nt = 0; nt < 4; ++nt) {
        const int c = ntile * 64 + nt * 16 + col;
#pragma unroll
        for (int i = 0; i < 8; ++i)
            out[((long)mstrip * 16 + rbase + i) * CD + c] = acc[nt][i];
    }
}

extern "C" void kernel_launch(void* const* d_in, const int* in_sizes, int n_in,
                              void* d_out, int out_size, void* d_ws, size_t ws_size,
                              hipStream_t stream) {
    const float* hs = (const float*)d_in[0];   // [B,S,D]
    const float* Wq = (const float*)d_in[1];   // [H,D,DH]
    const float* Wk = (const float*)d_in[2];
    const float* Wv = (const float*)d_in[3];
    const float* Wo = (const float*)d_in[4];   // [D,D]
    float* out = (float*)d_out;                // [B,S,D]

    char* ws = (char*)d_ws;
    size_t off = 0;
    __bf16* Xb  = (__bf16*)(ws + off); off += (size_t)CB * CS * CD * 2;        // 8 MB
    __bf16* Wqt = (__bf16*)(ws + off); off += (size_t)CH * CDH * CD * 2;       // 2 MB
    __bf16* Wkt = (__bf16*)(ws + off); off += (size_t)CH * CDH * CD * 2;
    __bf16* Wvt = (__bf16*)(ws + off); off += (size_t)CH * CDH * CD * 2;
    __bf16* Wot = (__bf16*)(ws + off); off += (size_t)CD * CD * 2;             // 2 MB
    __bf16* Qb  = (__bf16*)(ws + off); off += (size_t)CB * CH * CS * CDH * 2;  // 8 MB
    __bf16* Kb  = (__bf16*)(ws + off); off += (size_t)CB * CH * CS * CDH * 2;
    __bf16* Vtb = (__bf16*)(ws + off); off += (size_t)CB * CH * CDH * CS * 2;
    __bf16* Ccb = (__bf16*)(ws + off); off += (size_t)CB * CS * CD * 2;
    (void)ws_size; (void)in_sizes; (void)n_in; (void)out_size;

    const int nX = CB * CS * CD;                         // 4194304
    k_cvt<<<(nX + 255) / 256, 256, 0, stream>>>(hs, Xb, nX);

    const int nW = CH * CD * CDH;                        // 1048576
    k_tr_hde<<<(nW + 255) / 256, 256, 0, stream>>>(Wq, Wqt);
    k_tr_hde<<<(nW + 255) / 256, 256, 0, stream>>>(Wk, Wkt);
    k_tr_hde<<<(nW + 255) / 256, 256, 0, stream>>>(Wv, Wvt);
    const int nWo = CD * CD;
    k_tr_dd<<<(nWo + 255) / 256, 256, 0, stream>>>(Wo, Wot);

    // 4096 waves each for projection / attention / output, 8 waves per block
    k_qkv<<<512, 256, 0, stream>>>(Xb, Wqt, Wkt, Wvt, Qb, Kb, Vtb);
    k_attn<<<512, 256, 0, stream>>>(Qb, Kb, Vtb, Ccb);
    k_out<<<512, 256, 0, stream>>>(Ccb, Wot, out);
}
